// ModifiedFLCBcosConv2d_68109591381037
// MI455X (gfx1250) — matrix operand order repaired and checked by reading, not compiled, not run on hardware
//
#include <hip/hip_runtime.h>
#include <hip/hip_bf16.h>
#include <math.h>

// ---------------- problem constants ----------------
#define B_    32
#define CIN   256
#define HH    64
#define WW    64
#define COUT  256
#define KS    3
#define KTOT  (9 * 256)          // 2304, reordered as r*256 + ci
#define NPIX  (B_ * HH * WW)     // 131072 conv output pixels
#define OH    32
#define OW    32
#define TILE_PIX 18              // staged x-window [x0-1, x0+17)

static __device__ __forceinline__ float inv_scale() { return 100.0f / 48.0f; } // 1/SCALE
#define COS_EPS_  1e-6f
#define NORM_EPS_ 1e-6f

typedef __attribute__((ext_vector_type(8)))  __bf16 v8bf;
typedef __attribute__((ext_vector_type(16))) __bf16 v16bf;
typedef __attribute__((ext_vector_type(8)))  float  v8f;
typedef __attribute__((ext_vector_type(4)))  int    v4i;

#if __has_builtin(__builtin_amdgcn_global_load_async_to_lds_b128)
#define HAVE_ASYNC_LDS 1
typedef __attribute__((address_space(1))) v4i gv4i;
typedef __attribute__((address_space(3))) v4i lv4i;
#endif
#if __has_builtin(__builtin_amdgcn_s_wait_asynccnt)
#define HAVE_ASYNC_WAIT 1
#endif

// ---- A-operand fragment (contiguous k order, matching b_frag):
// lanes 0-15 hold K = {kc+0..15}; lanes 16-31 hold K = {kc+16..31}
__device__ __forceinline__ v16bf a_frag(const __bf16* kbase, int khalf) {
  union { v16bf v; struct { v8bf lo; v8bf hi; } s; } u;
  u.s.lo = *(const v8bf*)(kbase + khalf * 16);
  u.s.hi = *(const v8bf*)(kbase + khalf * 16 + 8);
  return u.v;
}

// ---- B-operand fragment: lane holds one column, 16 contiguous K at kc + khalf*16
__device__ __forceinline__ v16bf b_frag(const __bf16* colbase, int khalf) {
  return *(const v16bf*)(colbase + khalf * 16);
}

// =================== packing kernels ===================
__global__ __launch_bounds__(256) void pack_x_kernel(const float* __restrict__ x,
                                                     __bf16* __restrict__ xt) {
  long o = (long)blockIdx.x * 256 + threadIdx.x;   // NHWC output index, coalesced writes
  if (o >= (long)NPIX * CIN) return;
  int c  = (int)(o & 255);
  int xx = (int)((o >> 8) & 63);
  int yy = (int)((o >> 14) & 63);
  int b  = (int)(o >> 20);
  xt[o] = (__bf16)x[(((long)b * CIN + c) * HH + yy) * WW + xx];
}

__global__ __launch_bounds__(256) void pack_w_kernel(const float* __restrict__ w,
                                                     __bf16* __restrict__ wt) {
  long o = (long)blockIdx.x * 256 + threadIdx.x;   // wt[co][r*256+ci]
  if (o >= (long)COUT * KTOT) return;
  int ci = (int)(o & 255);
  int r  = (int)((o >> 8) % 9);
  int co = (int)(o / KTOT);
  wt[o] = (__bf16)w[((long)co * CIN + ci) * 9 + r];
}

__global__ __launch_bounds__(256) void sumsq_kernel(const __bf16* __restrict__ xt,
                                                    float* __restrict__ sq) {
  int p = blockIdx.x * 256 + threadIdx.x;
  if (p >= NPIX) return;
  const __bf16* row = xt + (long)p * CIN;
  float s = 0.f;
  for (int c = 0; c < CIN; c += 8) {
    v8bf v = *(const v8bf*)(row + c);
#pragma unroll
    for (int i = 0; i < 8; ++i) { float f = (float)v[i]; s += f * f; }
  }
  sq[p] = s;
}

__global__ __launch_bounds__(256) void norm_kernel(const float* __restrict__ sq,
                                                   float* __restrict__ nrm) {
  int p = blockIdx.x * 256 + threadIdx.x;
  if (p >= NPIX) return;
  int xx = p & 63, yy = (p >> 6) & 63, b = p >> 12;
  float s = 0.f;
#pragma unroll
  for (int dy = -1; dy <= 1; ++dy)
#pragma unroll
    for (int dx = -1; dx <= 1; ++dx) {
      int ys = yy + dy, xs = xx + dx;
      if (ys >= 0 && ys < HH && xs >= 0 && xs < WW)
        s += sq[((b * HH + ys) * WW) + xs];
    }
  nrm[p] = sqrtf(s + NORM_EPS_);
}

// FLC operator matrix  A[n,m] = (1/64) * sum_{k=-16..15} e^{2*pi*i*k*(2n-m)/64}
__global__ __launch_bounds__(256) void dft_kernel(__bf16* __restrict__ Arb,
                                                  __bf16* __restrict__ Aib) {
  int t = blockIdx.x * 256 + threadIdx.x;
  if (t >= 32 * 64) return;
  int n = t >> 6, m = t & 63;
  float cr = 0.f, ci = 0.f;
  float base = (2.0f * 3.14159265358979323846f / 64.0f) * (float)(2 * n - m);
  for (int k = -16; k <= 15; ++k) {
    float ang = base * (float)k;
    cr += __cosf(ang);
    ci += __sinf(ang);
  }
  Arb[t] = (__bf16)(cr * (1.0f / 64.0f));
  Aib[t] = (__bf16)(ci * (1.0f / 64.0f));
}

// =================== conv implicit GEMM (WMMA bf16, async-LDS staged) ===================
// One block = M-tile of 16 pixels (same b,y). 8 waves cover N=256 couts (two
// 16x16 accumulators each). Per dy row, the 18x256 bf16 activation window is
// staged once into LDS via global_load_async_to_lds_b128; all 8 waves and all
// 3 dx taps read A-fragments from LDS, weights stream from global (L2-resident).
__global__ __launch_bounds__(256) void conv_wmma_kernel(const __bf16* __restrict__ xt,
                                                        const __bf16* __restrict__ wt,
                                                        const float* __restrict__ nrm,
                                                        __bf16* __restrict__ convbuf) {
  __shared__ __bf16 atile[TILE_PIX * CIN];   // 9216 B

  const int lane   = threadIdx.x & 31;
  const int wave   = threadIdx.x >> 5;
  const int lane15 = lane & 15;
  const int khalf  = lane >> 4;

  const int p0 = blockIdx.x * 16;
  const int b  = p0 >> 12;
  const int y  = (p0 >> 6) & 63;
  const int x0 = p0 & 63;

  const int co0 = wave * 32 + lane15;

  v8f acc0 = {0.f, 0.f, 0.f, 0.f, 0.f, 0.f, 0.f, 0.f};
  v8f acc1 = {0.f, 0.f, 0.f, 0.f, 0.f, 0.f, 0.f, 0.f};

  for (int dy = -1; dy <= 1; ++dy) {
    const int ys = y + dy;
    const bool yok = (ys >= 0) && (ys < HH);

    // ---- stage the 18-pixel activation window for this row into LDS ----
    if (!yok) {
      for (int i = threadIdx.x; i < TILE_PIX * CIN / 2; i += 256)
        ((unsigned int*)atile)[i] = 0u;                       // whole row padded
    } else {
      const int xs0 = x0 - 1;
      const int lo = xs0 < 0 ? 0 : xs0;
      const int hi = (xs0 + TILE_PIX) > WW ? WW : (xs0 + TILE_PIX);
      if (xs0 < 0)                                            // left edge pad slot
        for (int i = threadIdx.x; i < CIN / 2; i += 256)
          ((unsigned int*)atile)[i] = 0u;
      if (xs0 + TILE_PIX > WW)                                // right edge pad slot
        for (int i = threadIdx.x; i < CIN / 2; i += 256)
          ((unsigned int*)atile)[(TILE_PIX - 1) * (CIN / 2) + i] = 0u;
      const __bf16* src = xt + (long)((b * HH + ys) * WW + lo) * CIN;
      __bf16*       dst = atile + (lo - xs0) * CIN;
      const int nvec = ((hi - lo) * CIN) / 8;                 // 16B beats
#if defined(HAVE_ASYNC_LDS)
      {
        gv4i* g = (gv4i*)src;
        lv4i* l = (lv4i*)dst;
        for (int i = threadIdx.x; i < nvec; i += 256)
          __builtin_amdgcn_global_load_async_to_lds_b128(g + i, l + i, 0, 0);
      }
#if defined(HAVE_ASYNC_WAIT)
      __builtin_amdgcn_s_wait_asynccnt(0);
#endif
#else
      for (int i = threadIdx.x; i < nvec; i += 256)
        ((v8bf*)dst)[i] = ((const v8bf*)src)[i];
#endif
    }
    __syncthreads();

    // ---- 3 dx taps x 8 ci-chunks, 2 WMMAs each ----
#pragma unroll
    for (int dx = -1; dx <= 1; ++dx) {
      const int r = (dy + 1) * 3 + (dx + 1);
      const __bf16* arow  = atile + (lane15 + dx + 1) * CIN;  // LDS reads (ds_load_b128)
      const __bf16* wrow0 = wt + (long)co0 * KTOT + r * CIN;
      const __bf16* wrow1 = wrow0 + (long)16 * KTOT;
      if (r < 8) {
        __builtin_prefetch(wrow0 + CIN, 0, 3);
        __builtin_prefetch(wrow1 + CIN, 0, 3);
      }
#pragma unroll
      for (int cc = 0; cc < 8; ++cc) {
        v16bf a  = a_frag(arow + cc * 32, khalf);
        v16bf b0 = b_frag(wrow0 + cc * 32, khalf);
        v16bf b1 = b_frag(wrow1 + cc * 32, khalf);
        acc0 = __builtin_amdgcn_wmma_f32_16x16x32_bf16(false, a, false, b0,
                                                       (short)0, acc0, false, false);
        acc1 = __builtin_amdgcn_wmma_f32_16x16x32_bf16(false, a, false, b1,
                                                       (short)0, acc1, false, false);
      }
    }
    __syncthreads();   // before next row overwrites the tile
  }

  // B-cos epilogue: out * (|out|/norm + eps) / SCALE, write NCHW bf16
#pragma unroll
  for (int v = 0; v < 8; ++v) {
    const int m  = khalf * 8 + v;       // D layout: M = khalf*8 + vgpr, N = lane15
    const int p  = p0 + m;
    const int xo = x0 + m;
    const float inv = 1.0f / nrm[p];
    float v0 = acc0[v];
    v0 = v0 * (fabsf(v0) * inv + COS_EPS_) * inv_scale();
    convbuf[(((long)b * COUT + co0) * HH + y) * WW + xo] = (__bf16)v0;
    float v1 = acc1[v];
    v1 = v1 * (fabsf(v1) * inv + COS_EPS_) * inv_scale();
    convbuf[(((long)b * COUT + co0 + 16) * HH + y) * WW + xo] = (__bf16)v1;
  }
}

// =================== FLC pool as two WMMA GEMMs per image ===================
// out = Ar*Y*Ar^T - Ai*Y*Ai^T. Stage1: P=Y*Ar^T, Q=Y*Ai^T stored TRANSPOSED in
// LDS (negating Q), stage2: out = Ar*P + Ai*(-Q).  One block per (b,co) image.
__global__ __launch_bounds__(256) void flc_wmma_kernel(const __bf16* __restrict__ convbuf,
                                                       const __bf16* __restrict__ Arb,
                                                       const __bf16* __restrict__ Aib,
                                                       float* __restrict__ out) {
  __shared__ __bf16 Pt[32 * 64];   // [v][y]
  __shared__ __bf16 Qt[32 * 64];   // [v][y], holds -Q

  const int lane   = threadIdx.x & 31;
  const int wave   = threadIdx.x >> 5;
  const int lane15 = lane & 15;
  const int khalf  = lane >> 4;
  const long img   = blockIdx.x;                    // b*COUT + co
  const __bf16* Y  = convbuf + img * (HH * WW);

  {
    const int ytile = wave >> 1;
    const int vtile = wave & 1;
    const int yrow  = ytile * 16 + lane15;
    const int vcol  = vtile * 16 + lane15;
    v8f accP = {0.f, 0.f, 0.f, 0.f, 0.f, 0.f, 0.f, 0.f};
    v8f accQ = {0.f, 0.f, 0.f, 0.f, 0.f, 0.f, 0.f, 0.f};
#pragma unroll
    for (int kc = 0; kc < 64; kc += 32) {
      v16bf a  = a_frag(Y + yrow * 64 + kc, khalf);
      v16bf br = b_frag(Arb + vcol * 64 + kc, khalf);
      v16bf bi = b_frag(Aib + vcol * 64 + kc, khalf);
      accP = __builtin_amdgcn_wmma_f32_16x16x32_bf16(false, a, false, br,
                                                     (short)0, accP, false, false);
      accQ = __builtin_amdgcn_wmma_f32_16x16x32_bf16(false, a, false, bi,
                                                     (short)0, accQ, false, false);
    }
#pragma unroll
    for (int e = 0; e < 8; ++e) {
      const int yy = ytile * 16 + khalf * 8 + e;
      const int vv = vtile * 16 + lane15;
      Pt[vv * 64 + yy] = (__bf16)accP[e];
      Qt[vv * 64 + yy] = (__bf16)(-accQ[e]);        // negate here: out -= Ai*Y*Ai^T
    }
  }
  __syncthreads();

  if (wave < 4) {
    const int utile = wave >> 1;
    const int vtile = wave & 1;
    const int urow  = utile * 16 + lane15;
    const int vcol  = vtile * 16 + lane15;
    v8f acc = {0.f, 0.f, 0.f, 0.f, 0.f, 0.f, 0.f, 0.f};
#pragma unroll
    for (int kc = 0; kc < 64; kc += 32) {
      v16bf aR = a_frag(Arb + urow * 64 + kc, khalf);
      v16bf aI = a_frag(Aib + urow * 64 + kc, khalf);
      v16bf bP = *(const v16bf*)(&Pt[vcol * 64 + kc + khalf * 16]);
      v16bf bQ = *(const v16bf*)(&Qt[vcol * 64 + kc + khalf * 16]);
      acc = __builtin_amdgcn_wmma_f32_16x16x32_bf16(false, aR, false, bP,
                                                    (short)0, acc, false, false);
      acc = __builtin_amdgcn_wmma_f32_16x16x32_bf16(false, aI, false, bQ,
                                                    (short)0, acc, false, false);
    }
#pragma unroll
    for (int e = 0; e < 8; ++e) {
      const int uu = utile * 16 + khalf * 8 + e;
      const int vv = vtile * 16 + lane15;
      out[img * (OH * OW) + uu * OW + vv] = acc[e];
    }
  }
}

// =================== launcher ===================
extern "C" void kernel_launch(void* const* d_in, const int* in_sizes, int n_in,
                              void* d_out, int out_size, void* d_ws, size_t ws_size,
                              hipStream_t stream) {
  const float* x = (const float*)d_in[0];
  const float* w = (const float*)d_in[1];
  float* out = (float*)d_out;

  char* ws = (char*)d_ws;
  size_t off = 0;
  auto take = [&](size_t bytes) { void* p = ws + off; off = (off + bytes + 255) & ~(size_t)255; return p; };

  __bf16* xt      = (__bf16*)take((size_t)NPIX * CIN * 2);          // 67 MB NHWC bf16
  __bf16* wt      = (__bf16*)take((size_t)COUT * KTOT * 2);         // 1.2 MB packed weights
  __bf16* convbuf = (__bf16*)take((size_t)B_ * COUT * HH * WW * 2); // 67 MB NCHW bf16 conv result
  float*  sq      = (float*)take((size_t)NPIX * 4);
  float*  nrm     = (float*)take((size_t)NPIX * 4);
  __bf16* Arb     = (__bf16*)take((size_t)32 * 64 * 2);
  __bf16* Aib     = (__bf16*)take((size_t)32 * 64 * 2);
  (void)ws_size; (void)in_sizes; (void)n_in; (void)out_size;

  pack_x_kernel<<<(NPIX * CIN) / 256, 256, 0, stream>>>(x, xt);
  pack_w_kernel<<<(COUT * KTOT + 255) / 256, 256, 0, stream>>>(w, wt);
  sumsq_kernel<<<NPIX / 256, 256, 0, stream>>>(xt, sq);
  norm_kernel<<<NPIX / 256, 256, 0, stream>>>(sq, nrm);
  dft_kernel<<<(32 * 64 + 255) / 256, 256, 0, stream>>>(Arb, Aib);
  conv_wmma_kernel<<<NPIX / 16, 256, 0, stream>>>(xt, wt, nrm, convbuf);
  flc_wmma_kernel<<<B_ * COUT, 256, 0, stream>>>(convbuf, Arb, Aib, out);
}
